// NoInputMassConserving_82231443849902
// MI455X (gfx1250) — matrix operand order repaired and checked by reading, not compiled, not run on hardware
//
#include <hip/hip_runtime.h>
#include <cstdint>
#include <cstddef>

// Problem constants (fixed by the reference).
#define B_    1024
#define T_    200
#define H_    64
#define AUX_  9
#define KPAD  128     // K=100 padded to 128 for bf16 WMMA (4 k-steps of 32)
#define ROWS  16      // batch rows per block (WMMA M)
#define NW    8       // waves per block
#define NT_   (NW * 32)

typedef __attribute__((ext_vector_type(16))) __bf16 v16bf;
typedef __attribute__((ext_vector_type(8)))  float  v8f;

union Frag {
  uint4 q[2];   // 32 bytes = 16 bf16
  v16bf v;
};

__device__ __forceinline__ unsigned short f2bf(float x) {
  unsigned int u = __float_as_uint(x);
  u += 0x7FFFu + ((u >> 16) & 1u);          // round-to-nearest-even
  return (unsigned short)(u >> 16);
}

// One-time repack: Wr2 (100 x 4096, f32, row-major) -> wr2t (4096 x 128, bf16,
// N-major, K zero-padded).  1 MB -> lives in L2 for the whole scan.
__global__ __launch_bounds__(256) void repack_wr2(const float* __restrict__ Wr2,
                                                  unsigned short* __restrict__ wr2t) {
  int idx = blockIdx.x * 256 + threadIdx.x;
  if (idx >= 4096 * KPAD) return;
  int n = idx >> 7;          // output column of the GEMM
  int k = idx & (KPAD - 1);
  float v = (k < 100) ? Wr2[k * 4096 + n] : 0.0f;
  wr2t[idx] = f2bf(v);
}

__global__ __launch_bounds__(NT_, 1) void fused_scan(
    const float* __restrict__ init_state,
    const float* __restrict__ xa,
    const float* __restrict__ W1,  const float* __restrict__ b1,
    const float* __restrict__ W2,  const float* __restrict__ b2,
    const float* __restrict__ Wr1, const float* __restrict__ br1,
    const float* __restrict__ br2,
    const unsigned short* __restrict__ wr2t,
    float* __restrict__ m_out, float* __restrict__ c_arr,
    float* __restrict__ r_arr)
{
  // ---- LDS (~171 KB; CDNA5 allows up to 320 KB per workgroup) ----
  __shared__ float W1s[73 * 50];
  __shared__ float b1s[50];
  __shared__ float W2s[50 * 100];
  __shared__ float b2s[100];
  __shared__ float Wr1s[100 * 100];
  __shared__ float br1s[100];
  __shared__ float br2s[4096];
  __shared__ float cts[ROWS][H_];            // recurrent state (16 batch rows)
  __shared__ float xs[ROWS][AUX_];
  __shared__ float h1s[ROWS][50];
  __shared__ float h2s[ROWS][100];
  __shared__ unsigned short rhb[ROWS][KPAD]; // tanh output, bf16, K-padded
  __shared__ float Lg[NW][ROWS][64];         // per-wave logits/probs tile
  __shared__ float accs[NW][ROWS][64];       // per-wave c_out partial sums

  const int tid = threadIdx.x;
  const int b0  = blockIdx.x * ROWS;
  const int ln  = tid & 31;
  const int w   = tid >> 5;
  const int lm  = ln & 15;        // M (A) / N (B) lane index
  const int hi  = ln >> 4;        // half-wave selector

  // ---- stage weights into LDS (one time; L2-hot) ----
  for (int i = tid; i < 73 * 50;  i += NT_) W1s[i]  = W1[i];
  for (int i = tid; i < 50;       i += NT_) b1s[i]  = b1[i];
  for (int i = tid; i < 50 * 100; i += NT_) W2s[i]  = W2[i];
  for (int i = tid; i < 100;      i += NT_) { b2s[i] = b2[i]; br1s[i] = br1[i]; }
  for (int i = tid; i < 100 * 100; i += NT_) Wr1s[i] = Wr1[i];
  for (int i = tid; i < 4096;     i += NT_) br2s[i] = br2[i];
  for (int i = tid; i < ROWS * H_; i += NT_) {
    int r = i >> 6, j = i & 63;
    float v = init_state[(size_t)(b0 + r) * H_ + j];
    cts[r][j] = v;
    c_arr[(size_t)(b0 + r) * ((T_ + 1) * H_) + j] = v;   // c[:,0,:] = init_state
  }
  for (int i = tid; i < NW * ROWS * 64; i += NT_) (&accs[0][0][0])[i] = 0.0f;
  for (int i = tid; i < ROWS * (KPAD - 100); i += NT_) {  // zero K padding once
    int r = i / (KPAD - 100), c = 100 + i % (KPAD - 100);
    rhb[r][c] = 0;
  }
  __syncthreads();

  for (int t = 0; t < T_; ++t) {
    // ---- load xa[:, t, :] tile ----
    if (tid < ROWS * AUX_) {
      int r = tid / AUX_, a = tid % AUX_;
      xs[r][a] = xa[(size_t)(b0 + r) * (T_ * AUX_) + t * AUX_ + a];
    }
    __syncthreads();

    // ---- tiny MLP in f32 VALU (19 MMAC/step total: negligible vs GEMM) ----
    for (int idx = tid; idx < ROWS * 50; idx += NT_) {          // layer 1 (relu)
      int r = idx / 50, c = idx % 50;
      float s = b1s[c];
      #pragma unroll 8
      for (int k = 0; k < H_; ++k) s = fmaf(cts[r][k], W1s[k * 50 + c], s);
      #pragma unroll
      for (int k = 0; k < AUX_; ++k) s = fmaf(xs[r][k], W1s[(H_ + k) * 50 + c], s);
      h1s[r][c] = fmaxf(s, 0.0f);
    }
    __syncthreads();
    for (int idx = tid; idx < ROWS * 100; idx += NT_) {         // layer 2 (relu)
      int r = idx / 100, c = idx % 100;
      float s = b2s[c];
      #pragma unroll 10
      for (int k = 0; k < 50; ++k) s = fmaf(h1s[r][k], W2s[k * 100 + c], s);
      h2s[r][c] = fmaxf(s, 0.0f);
    }
    __syncthreads();
    for (int idx = tid; idx < ROWS * 100; idx += NT_) {         // layer 3 (tanh) -> bf16
      int r = idx / 100, c = idx % 100;
      float s = br1s[c];
      #pragma unroll 10
      for (int k = 0; k < 100; ++k) s = fmaf(h2s[r][k], Wr1s[k * 100 + c], s);
      rhb[r][c] = f2bf(tanhf(s));
    }
    __syncthreads();

    // ---- A fragments (16x128 bf16), per the CDNA5 16-bit A layout:
    // lanes 0-15 (row m): K {kb..kb+7, kb+16..kb+23}; lanes 16-31: K {+8, +24}.
    Frag afr[4];
    #pragma unroll
    for (int kk = 0; kk < 4; ++kk) {
      int kb = kk * 32;
      afr[kk].q[0] = *(const uint4*)&rhb[lm][kb + hi * 8];
      afr[kk].q[1] = *(const uint4*)&rhb[lm][kb + 16 + hi * 8];
    }

    // ---- fused GEMM (M16,N64,K128 per group) + softmax + weighted reduce.
    // Wave w owns softmax groups i = w, w+8, ..., w+56: no cross-wave barriers.
    for (int g = 0; g < 64 / NW; ++g) {
      const int i = g * NW + w;

      #pragma unroll
      for (int nt = 0; nt < 4; ++nt) {
        const int rowN = i * 64 + nt * 16 + lm;
        const unsigned short* bp = wr2t + (size_t)rowN * KPAD + (hi << 3);
        v8f acc = {0.f, 0.f, 0.f, 0.f, 0.f, 0.f, 0.f, 0.f};
        #pragma unroll
        for (int kk = 0; kk < 4; ++kk) {
          Frag bf;
          bf.q[0] = *(const uint4*)(bp + kk * 32);
          bf.q[1] = *(const uint4*)(bp + kk * 32 + 16);
          acc = __builtin_amdgcn_wmma_f32_16x16x32_bf16(
              false, afr[kk].v, false, bf.v, (short)0, acc, false, false);
        }
        // C layout: VGPR l -> row l (lanes 0-15) / row l+8 (lanes 16-31), col = lane%16
        #pragma unroll
        for (int l = 0; l < 8; ++l) {
          int rr = l + hi * 8;
          int cc = nt * 16 + lm;
          Lg[w][rr][cc] = acc[l] + br2s[i * 64 + cc];
        }
      }

      // online softmax over the 64 columns of this group; 2 lanes per row
      const int r  = ln >> 1;
      const int cb = (ln & 1) * 32;
      float mx = -3.4e38f;
      #pragma unroll 8
      for (int c = 0; c < 32; ++c) mx = fmaxf(mx, Lg[w][r][cb + c]);
      mx = fmaxf(mx, __shfl_xor(mx, 1));
      float s = 0.0f;
      #pragma unroll 8
      for (int c = 0; c < 32; ++c) {
        float e = __expf(Lg[w][r][cb + c] - mx);
        Lg[w][r][cb + c] = e;
        s += e;
      }
      s += __shfl_xor(s, 1);
      const float wgt = cts[r][i] / s;                  // ct[b,i] * softmax row i
      #pragma unroll 8
      for (int c = 0; c < 32; ++c)
        accs[w][r][cb + c] += Lg[w][r][cb + c] * wgt;

      // rdiag[k] = r[0, 63-k, k]  ->  group i contributes k = 63-i (batch row 0)
      if (b0 == 0 && r == 0) {
        int jd = 63 - i;
        if ((jd >> 5) == (ln & 1))
          r_arr[(size_t)t * 64 + jd] = Lg[w][0][jd] / s;
      }
    }
    __syncthreads();

    // ---- combine per-wave partials, emit outputs, update state ----
    for (int idx = tid; idx < ROWS * H_; idx += NT_) {
      int r = idx >> 6, j = idx & 63;
      float v = 0.0f;
      #pragma unroll
      for (int ww = 0; ww < NW; ++ww) { v += accs[ww][r][j]; accs[ww][r][j] = 0.0f; }
      m_out[(size_t)(b0 + r) * (T_ * H_) + (size_t)t * H_ + j] = v;
      c_arr[(size_t)(b0 + r) * ((T_ + 1) * H_) + (size_t)(t + 1) * H_ + j] =
          1.0002f * v - 0.0001f;
      cts[r][j] = v;
    }
    __syncthreads();
  }
}

extern "C" void kernel_launch(void* const* d_in, const int* in_sizes, int n_in,
                              void* d_out, int out_size, void* d_ws, size_t ws_size,
                              hipStream_t stream) {
  (void)in_sizes; (void)n_in; (void)out_size; (void)ws_size;
  const float* init_state = (const float*)d_in[0];
  // d_in[1] = n_time_steps (scalar 200; compile-time constant here)
  const float* xa  = (const float*)d_in[2];
  const float* W1  = (const float*)d_in[3];
  const float* b1  = (const float*)d_in[4];
  const float* W2  = (const float*)d_in[5];
  const float* b2  = (const float*)d_in[6];
  const float* Wr1 = (const float*)d_in[7];
  const float* br1 = (const float*)d_in[8];
  const float* Wr2 = (const float*)d_in[9];
  const float* br2 = (const float*)d_in[10];

  float* out   = (float*)d_out;
  float* m_out = out;                                        // (B, T, H)
  float* c_arr = out + (size_t)B_ * T_ * H_;                 // (B, T+1, H)
  float* r_arr = c_arr + (size_t)B_ * (T_ + 1) * H_;         // (1, T, 64)

  unsigned short* wr2t = (unsigned short*)d_ws;              // 4096*128 bf16 = 1 MB

  repack_wr2<<<(4096 * KPAD + 255) / 256, 256, 0, stream>>>(Wr2, wr2t);
  fused_scan<<<B_ / ROWS, NT_, 0, stream>>>(init_state, xa, W1, b1, W2, b2,
                                            Wr1, br1, br2, wr2t,
                                            m_out, c_arr, r_arr);
}
